// LlamaAttention_27908697490161
// MI455X (gfx1250) — compile-verified, matrix operand order checked by reading, not run on hardware
//
#include <hip/hip_runtime.h>
#include <hip/hip_bf16.h>

// ---------------------------------------------------------------------------
// CDNA5 (gfx1250) Llama attention block, bf16 WMMA + async global->LDS.
// B=2, S=2048, E=4096, H=32, D=128, T=B*S=4096.
// ---------------------------------------------------------------------------

typedef __attribute__((ext_vector_type(16))) __bf16 v16bf;
typedef __attribute__((ext_vector_type(8)))  __bf16 v8bf;
typedef __attribute__((ext_vector_type(2)))  __bf16 v2bf;
typedef __attribute__((ext_vector_type(8)))  float  v8f;

union Frag16 { v16bf v; v8bf h[2]; };

__device__ __forceinline__ unsigned short f2bf_bits(float f) {
    union { float f; unsigned int u; } v; v.f = f;
    unsigned int u = v.u;
    u += 0x7fffu + ((u >> 16) & 1u);           // round-to-nearest-even
    return (unsigned short)(u >> 16);
}
__device__ __forceinline__ float bf2f(unsigned short b) {
    union { unsigned int u; float f; } v; v.u = ((unsigned int)b) << 16;
    return v.f;
}
__device__ __forceinline__ unsigned pack2_bf16(float a, float b) {
#if __has_builtin(__builtin_amdgcn_cvt_pk_bf16_f32)
    union { v2bf v; unsigned u; } r;
    r.v = __builtin_amdgcn_cvt_pk_bf16_f32(a, b);
    return r.u;
#else
    return (unsigned)f2bf_bits(a) | ((unsigned)f2bf_bits(b) << 16);
#endif
}

__device__ __forceinline__ v8f wmma_bf16(const Frag16& a, const Frag16& b, v8f c) {
    return __builtin_amdgcn_wmma_f32_16x16x32_bf16(
        false, a.v, false, b.v, (short)0, c, false, false);
}

// ---- CDNA5 async global->LDS copy (ASYNCcnt-tracked DMA path) -------------
__device__ __forceinline__ void async_load_b128(unsigned lds_off, const void* g) {
    asm volatile("global_load_async_to_lds_b128 %0, %1, off"
                 :: "v"(lds_off), "v"((unsigned long long)(size_t)g)
                 : "memory");
}
template <int N>
__device__ __forceinline__ void wait_asynccnt_le() {
    asm volatile("s_wait_asynccnt %0" :: "n"(N) : "memory");
}

// ---------------------------------------------------------------------------
// f32 -> bf16 conversion
// ---------------------------------------------------------------------------
__global__ void cvt_f32_bf16(const float* __restrict__ in,
                             unsigned short* __restrict__ out, int n) {
    int i = (blockIdx.x * blockDim.x + threadIdx.x) * 4;
    if (i + 3 < n) {
        float4 f = *(const float4*)&in[i];
        unsigned* o = (unsigned*)&out[i];
        o[0] = pack2_bf16(f.x, f.y);
        o[1] = pack2_bf16(f.z, f.w);
    } else {
        for (int j = i; j < n; ++j) out[j] = f2bf_bits(in[j]);
    }
}

// ---------------------------------------------------------------------------
// GEMM: C[M,N] = A[M,K] * Bw[N,K]^T   (bf16 operands, K contiguous)
// OUT_MODE 0: bf16 row-major [M,N]
// OUT_MODE 1: bf16 per-head transposed: (m,n) -> [b][h][d][s]  (for V)
// OUT_MODE 2: f32 row-major [M,N]
// Block tile 128x128, BK=64, 8 waves (4x2), double-buffered async LDS fill.
// ---------------------------------------------------------------------------
template <int OUT_MODE>
__global__ __launch_bounds__(256) void gemm_xwt(
    const unsigned short* __restrict__ A,
    const unsigned short* __restrict__ Bw,
    void* __restrict__ out, int M, int N, int K)
{
    constexpr int BM = 128, BN = 128, BK = 64, LDT = 72; // 72: bank-phase pad
    __shared__ unsigned short lA[2][BM * LDT];
    __shared__ unsigned short lB[2][BN * LDT];

    const int tid  = threadIdx.x;
    const int lane = tid & 31;
    const int wave = tid >> 5;
    const int l15  = lane & 15;
    const int hi   = lane >> 4;

    const int m0 = blockIdx.y * BM;
    const int n0 = blockIdx.x * BN;
    const int mw = (wave >> 1) * 32;
    const int nw = (wave & 1) * 64;

    v8f acc[2][4];
#pragma unroll
    for (int mt = 0; mt < 2; ++mt)
#pragma unroll
        for (int nt = 0; nt < 4; ++nt) acc[mt][nt] = (v8f)(0.0f);

    auto load_tile = [&](int k0, int buf) {
#pragma unroll
        for (int i = 0; i < 4; ++i) {            // 1024 chunks / 256 threads
            int c   = tid + i * 256;
            int row = c >> 3;                    // 8 x 16B chunks per row
            int col = (c & 7) * 8;
            async_load_b128((unsigned)(size_t)&lA[buf][row * LDT + col],
                            &A[(size_t)(m0 + row) * K + k0 + col]);
            async_load_b128((unsigned)(size_t)&lB[buf][row * LDT + col],
                            &Bw[(size_t)(n0 + row) * K + k0 + col]);
        }
    };

    load_tile(0, 0);
    int cur = 0;
    for (int k0 = 0; k0 < K; k0 += BK) {
        const bool has_next = (k0 + BK) < K;
        if (has_next) load_tile(k0 + BK, cur ^ 1);
        if (has_next) wait_asynccnt_le<8>(); else wait_asynccnt_le<0>();
        __syncthreads();

#pragma unroll
        for (int kk = 0; kk < 2; ++kk) {
            Frag16 af[2], bf_[4];
#pragma unroll
            for (int mt = 0; mt < 2; ++mt) {
                int r = mw + mt * 16 + l15;
                af[mt].h[0] = *(const v8bf*)&lA[cur][r * LDT + kk * 32 + hi * 8];
                af[mt].h[1] = *(const v8bf*)&lA[cur][r * LDT + kk * 32 + 16 + hi * 8];
            }
#pragma unroll
            for (int nt = 0; nt < 4; ++nt) {
                int r = nw + nt * 16 + l15;
                bf_[nt].h[0] = *(const v8bf*)&lB[cur][r * LDT + kk * 32 + hi * 8];
                bf_[nt].h[1] = *(const v8bf*)&lB[cur][r * LDT + kk * 32 + 16 + hi * 8];
            }
#pragma unroll
            for (int mt = 0; mt < 2; ++mt)
#pragma unroll
                for (int nt = 0; nt < 4; ++nt)
                    acc[mt][nt] = wmma_bf16(af[mt], bf_[nt], acc[mt][nt]);
        }
        __syncthreads();
        cur ^= 1;
    }

#pragma unroll
    for (int mt = 0; mt < 2; ++mt)
#pragma unroll
        for (int nt = 0; nt < 4; ++nt)
#pragma unroll
            for (int r = 0; r < 8; ++r) {
                int m = m0 + mw + mt * 16 + r + 8 * hi;
                int n = n0 + nw + nt * 16 + l15;
                float val = acc[mt][nt][r];
                if constexpr (OUT_MODE == 0) {
                    ((unsigned short*)out)[(size_t)m * N + n] = f2bf_bits(val);
                } else if constexpr (OUT_MODE == 1) {
                    int b = m >> 11, s = m & 2047;     // S = 2048
                    int h = n >> 7,  d = n & 127;      // D = 128
                    ((unsigned short*)out)[(size_t)((b * 32 + h) * 128 + d) * 2048 + s] =
                        f2bf_bits(val);
                } else {
                    ((float*)out)[(size_t)m * N + n] = val;
                }
            }
}

// ---------------------------------------------------------------------------
// RoPE (reference rotation is position-independent: angles depend on (h, i))
// ---------------------------------------------------------------------------
__global__ void rope_kernel(unsigned short* __restrict__ q,
                            unsigned short* __restrict__ k,
                            const float* __restrict__ cosp,
                            const float* __restrict__ sinp, int total)
{
    int idx = blockIdx.x * blockDim.x + threadIdx.x;
    if (idx >= total) return;                 // total = T*H*(D/2)
    int i = idx & 63;
    int h = (idx >> 6) & 31;
    int t = idx >> 11;
    float c = cosp[h * 64 + i];
    float s = sinp[h * 64 + i];
    size_t base = (size_t)t * 4096 + h * 128 + i * 2;
    float q0 = bf2f(q[base]), q1 = bf2f(q[base + 1]);
    ((unsigned*)q)[base >> 1] = pack2_bf16(q0 * c - q1 * s, q0 * s + q1 * c);
    float k0 = bf2f(k[base]), k1 = bf2f(k[base + 1]);
    ((unsigned*)k)[base >> 1] = pack2_bf16(k0 * c - k1 * s, k0 * s + k1 * c);
}

// ---------------------------------------------------------------------------
// Flash attention v2: grid (S/128, H, B), 8 waves. Each wave owns a 16-row Q
// tile; the whole workgroup shares each 32-key K-tile and V-tile, staged into
// double-buffered LDS with global_load_async_to_lds_b128 (prefetch tile i+1
// while computing tile i). Fragments come from LDS (short-latency ds_read)
// => low register pressure, pipelined WMMA.
//   qb, kb: bf16 [t][h*128+d] (post-RoPE);  vt: bf16 [b][h][d][s]
// ---------------------------------------------------------------------------
__global__ __launch_bounds__(256) void attn_kernel(
    const unsigned short* __restrict__ qb,
    const unsigned short* __restrict__ kb,
    const unsigned short* __restrict__ vt,
    unsigned short* __restrict__ ctx)
{
    constexpr int LDK = 136;   // K-tile row stride (32 rows x 128 d), 4-bank phase
    constexpr int LDV = 40;    // V-tile row stride (128 d rows x 32 keys)
    __shared__ unsigned short kTile[2][32 * LDK];
    __shared__ unsigned short vTile[2][128 * LDV];
    __shared__ float          sS[8][16][34];
    __shared__ unsigned short sP[8][16][40];
    __shared__ float          sCorr[8][16];

    const int tid  = threadIdx.x;
    const int lane = tid & 31;
    const int wave = tid >> 5;
    const int l15  = lane & 15;
    const int hi   = lane >> 4;
    const int b    = blockIdx.z;
    const int h    = blockIdx.y;
    const int q0row = blockIdx.x * 128 + wave * 16;

    const size_t kbase = (size_t)b * 2048 * 4096 + h * 128;               // kb[b*S + s][h*128]
    const size_t vbase = (size_t)((b * 32 + h) * 128) * 2048;             // vt[b][h][d][s]

    // stage one 32-key K tile + V tile into LDS buffer `buf` (async)
    auto load_tiles = [&](int kt, int buf) {
#pragma unroll
        for (int i = 0; i < 2; ++i) {
            // K tile: 32 rows x 256B  = 512 chunks
            int c   = tid + i * 256;
            int krow = c >> 4;                   // 16 x 16B chunks per row
            int kcol = (c & 15) * 8;
            async_load_b128((unsigned)(size_t)&kTile[buf][krow * LDK + kcol],
                            &kb[kbase + (size_t)(kt + krow) * 4096 + kcol]);
            // V tile: 128 rows x 64B = 512 chunks
            int vrow = c >> 2;                   // 4 x 16B chunks per row
            int vcol = (c & 3) * 8;
            async_load_b128((unsigned)(size_t)&vTile[buf][vrow * LDV + vcol],
                            &vt[vbase + (size_t)vrow * 2048 + kt + vcol]);
        }
    };

    // Q fragments (persistent, 4 chunks of 32 over d=128)
    Frag16 qf[4];
    {
        size_t qrow = ((size_t)(b * 2048 + q0row + l15)) * 4096 + h * 128;
#pragma unroll
        for (int kd = 0; kd < 4; ++kd) {
            qf[kd].h[0] = *(const v8bf*)&qb[qrow + kd * 32 + hi * 8];
            qf[kd].h[1] = *(const v8bf*)&qb[qrow + kd * 32 + 16 + hi * 8];
        }
    }

    v8f o[8];
#pragma unroll
    for (int d8 = 0; d8 < 8; ++d8) o[d8] = (v8f)(0.0f);
    float rmax = -3.0e38f;
    float rsum = 0.0f;
    const float scale = 0.08838834764831845f;  // 1/sqrt(128)

    load_tiles(0, 0);
    int cur = 0;
    for (int kt = 0; kt < 2048; kt += 32) {
        const bool has_next = (kt + 32) < 2048;
        if (has_next) load_tiles(kt + 32, cur ^ 1);
        if (has_next) wait_asynccnt_le<4>(); else wait_asynccnt_le<0>();
        __syncthreads();                       // tiles visible to all waves

        // ----- S = Q * K^T (16 x 32): B-fragments from LDS K tile ----------
        v8f sc[2];
#pragma unroll
        for (int nt = 0; nt < 2; ++nt) {
            sc[nt] = (v8f)(0.0f);
            const int krow = (nt * 16 + l15) * LDK;
#pragma unroll
            for (int kd = 0; kd < 4; ++kd) {
                Frag16 kf;
                kf.h[0] = *(const v8bf*)&kTile[cur][krow + kd * 32 + hi * 8];
                kf.h[1] = *(const v8bf*)&kTile[cur][krow + kd * 32 + 16 + hi * 8];
                sc[nt] = wmma_bf16(qf[kd], kf, sc[nt]);
            }
        }
        // ----- C-layout -> per-wave LDS scratch ---------------------------
#pragma unroll
        for (int nt = 0; nt < 2; ++nt)
#pragma unroll
            for (int r = 0; r < 8; ++r)
                sS[wave][r + 8 * hi][nt * 16 + l15] = sc[nt][r] * scale;

        // ----- online softmax (lane l: row l15, half hi) -------------------
        float tmax = -3.0e38f;
#pragma unroll
        for (int j = 0; j < 16; ++j)
            tmax = fmaxf(tmax, sS[wave][l15][hi * 16 + j]);
        tmax = fmaxf(tmax, __shfl_xor(tmax, 16, 32));
        float newmax = fmaxf(rmax, tmax);
        float corr   = __expf(rmax - newmax);
        rmax = newmax;

        float tsum = 0.0f;
#pragma unroll
        for (int j = 0; j < 16; ++j) {
            float p = __expf(sS[wave][l15][hi * 16 + j] - newmax);
            tsum += p;
            sP[wave][l15][hi * 16 + j] = f2bf_bits(p);
        }
        tsum += __shfl_xor(tsum, 16, 32);
        rsum = rsum * corr + tsum;

        if (hi == 0) sCorr[wave][l15] = corr;
        float cr[8];
#pragma unroll
        for (int r = 0; r < 8; ++r) cr[r] = sCorr[wave][r + 8 * hi];
#pragma unroll
        for (int d8 = 0; d8 < 8; ++d8)
#pragma unroll
            for (int r = 0; r < 8; ++r) o[d8][r] *= cr[r];

        // ----- O += P * V: B-fragments from LDS V tile ---------------------
        Frag16 pf;
        pf.h[0] = *(const v8bf*)&sP[wave][l15][hi * 8];
        pf.h[1] = *(const v8bf*)&sP[wave][l15][16 + hi * 8];
#pragma unroll
        for (int d8 = 0; d8 < 8; ++d8) {
            const int vrow = (d8 * 16 + l15) * LDV;
            Frag16 vf;
            vf.h[0] = *(const v8bf*)&vTile[cur][vrow + hi * 8];
            vf.h[1] = *(const v8bf*)&vTile[cur][vrow + 16 + hi * 8];
            o[d8] = wmma_bf16(pf, vf, o[d8]);
        }
        __syncthreads();                       // all reads done before overwrite
        cur ^= 1;
    }

    // ----- normalize + write context --------------------------------------
    if (hi == 0) sCorr[wave][l15] = 1.0f / rsum;
    float inv[8];
#pragma unroll
    for (int r = 0; r < 8; ++r) inv[r] = sCorr[wave][r + 8 * hi];
#pragma unroll
    for (int d8 = 0; d8 < 8; ++d8)
#pragma unroll
        for (int r = 0; r < 8; ++r) {
            int s = q0row + r + 8 * hi;
            size_t t = (size_t)b * 2048 + s;
            ctx[t * 4096 + h * 128 + d8 * 16 + l15] = f2bf_bits(o[d8][r] * inv[r]);
        }
}

// ---------------------------------------------------------------------------
// Host-side launch
// ---------------------------------------------------------------------------
extern "C" void kernel_launch(void* const* d_in, const int* in_sizes, int n_in,
                              void* d_out, int out_size, void* d_ws, size_t ws_size,
                              hipStream_t stream) {
    (void)in_sizes; (void)n_in; (void)out_size; (void)ws_size;
    const float* x    = (const float*)d_in[0];
    const float* cosp = (const float*)d_in[2];
    const float* sinp = (const float*)d_in[3];
    const float* wq   = (const float*)d_in[4];
    const float* wk   = (const float*)d_in[5];
    const float* wv   = (const float*)d_in[6];
    const float* wo   = (const float*)d_in[7];
    float* out = (float*)d_out;

    const size_t SZ = (size_t)4096 * 4096;
    unsigned short* xb  = (unsigned short*)d_ws;
    unsigned short* wqb = xb  + SZ;
    unsigned short* wkb = wqb + SZ;
    unsigned short* wvb = wkb + SZ;
    unsigned short* wob = wvb + SZ;
    unsigned short* qb  = wob + SZ;
    unsigned short* kb  = qb  + SZ;
    unsigned short* vt  = kb  + SZ;
    unsigned short* ctx = vt  + SZ;            // total 9 * 32MB = 288MB

    const int cvtGrid = (int)(SZ / (4 * 256));
    cvt_f32_bf16<<<cvtGrid, 256, 0, stream>>>(x,  xb,  (int)SZ);
    cvt_f32_bf16<<<cvtGrid, 256, 0, stream>>>(wq, wqb, (int)SZ);
    cvt_f32_bf16<<<cvtGrid, 256, 0, stream>>>(wk, wkb, (int)SZ);
    cvt_f32_bf16<<<cvtGrid, 256, 0, stream>>>(wv, wvb, (int)SZ);
    cvt_f32_bf16<<<cvtGrid, 256, 0, stream>>>(wo, wob, (int)SZ);

    dim3 gg(32, 32);
    gemm_xwt<0><<<gg, 256, 0, stream>>>(xb, wqb, qb, 4096, 4096, 4096);
    gemm_xwt<0><<<gg, 256, 0, stream>>>(xb, wkb, kb, 4096, 4096, 4096);
    gemm_xwt<1><<<gg, 256, 0, stream>>>(xb, wvb, vt, 4096, 4096, 4096);

    const int ropeTotal = 4096 * 32 * 64;      // T*H*(D/2)
    rope_kernel<<<ropeTotal / 256, 256, 0, stream>>>(qb, kb, cosp, sinp, ropeTotal);

    attn_kernel<<<dim3(16, 32, 2), 256, 0, stream>>>(qb, kb, vt, ctx);

    gemm_xwt<2><<<gg, 256, 0, stream>>>(ctx, wob, out, 4096, 4096, 4096);
}
